// RNN_Variational_Decoder_35837207118045
// MI455X (gfx1250) — matrix operand order repaired and checked by reading, not compiled, not run on hardware
//
#include <hip/hip_runtime.h>
#include <hip/hip_bf16.h>
#include <stdint.h>

// ---------------- problem constants (match reference) ----------------
#define BATCH 512
#define TMAX  200
#define HID   1024
#define DMLP  1024
#define DOUT  256
#define XP    (DOUT + 8)   // LDS row pitch for x  (+16B => 4-bank skew, conflict-free b128)
#define HP    (HID + 8)    // LDS row pitch for h / t1 / t2

// ---------------- vector types for WMMA ----------------
typedef __attribute__((ext_vector_type(16))) __bf16  v16bf;
typedef __attribute__((ext_vector_type(8)))  float   v8f;
typedef __attribute__((ext_vector_type(4)))  unsigned int u32x4;

union ABu { v16bf v; u32x4 q[2]; };

// A-matrix 16x32 bf16 tile from an LDS array (row-pitched). Per ISA layout:
// lane<16 : row=lane,   K = kb+[0..7]  (q0) and kb+16+[0..7]  (q1)
// lane>=16: row=lane-16, same with kb already offset by +8 by caller.
#define LOAD_A16(arr, row_, kb_) ({                                   \
  ABu u_;                                                             \
  u_.q[0] = *(const u32x4*)(&(arr)[(row_)][(kb_)]);                   \
  u_.q[1] = *(const u32x4*)(&(arr)[(row_)][(kb_) + 16]);              \
  u_.v; })

// B-matrix 32x16 bf16 tile from a row-major bf16 weight matrix W[NO][K]:
// lane n holds column n = row n of W; K = kb+[0..7] (q0) and kb+16+[0..7] (q1).
#define LOAD_B16(Wp_, rowN_, K_, kb_) ({                              \
  ABu u_;                                                             \
  const u32x4* p_ = (const u32x4*)((Wp_) + (size_t)(rowN_)*(K_) + (kb_)); \
  u_.q[0] = p_[0];                                                    \
  u_.q[1] = p_[2];                                                    \
  u_.v; })

#define WMMA_BF16(acc_, a_, b_)                                       \
  acc_ = __builtin_amdgcn_wmma_f32_16x16x32_bf16(false, (a_), false, (b_), \
                                                 (short)0, (acc_), false, false)

__device__ __forceinline__ unsigned short f2bf(float f) {
  unsigned int u = __float_as_uint(f);
  return (unsigned short)((u + 0x7FFFu + ((u >> 16) & 1u)) >> 16);  // RNE
}
__device__ __forceinline__ float sigf(float x)     { return 1.0f / (1.0f + __expf(-x)); }
__device__ __forceinline__ float tanhfast(float x) { return 1.0f - 2.0f / (__expf(2.0f * x) + 1.0f); }

// ---------------- packed-sequence setup: bs[t] and exclusive prefix offsets --------
__global__ void pack_setup_kernel(const int* __restrict__ lengths,
                                  int* __restrict__ bs, long* __restrict__ offs) {
  __shared__ int sbs[TMAX];
  const int t = threadIdx.x;
  if (t < TMAX) {
    int c = 0;
    for (int b = 0; b < BATCH; ++b) c += (lengths[b] > t) ? 1 : 0;
    sbs[t] = c;
    bs[t]  = c;
  }
  __syncthreads();
  if (t == 0) {
    long acc = 0;
    for (int i = 0; i < TMAX; ++i) { offs[i] = acc; acc += sbs[i]; }
  }
}

// ---------------- f32 -> bf16 weight conversion (once per launch) ----------------
__global__ void cvt_bf16_kernel(const float* __restrict__ src,
                                unsigned short* __restrict__ dst, int n) {
  for (int i = blockIdx.x * blockDim.x + threadIdx.x; i < n; i += gridDim.x * blockDim.x)
    dst[i] = f2bf(src[i]);
}

// ---------------- fused T-step variational decoder ----------------
// One workgroup owns 16 batch rows for the whole recurrence; 8 waves split the
// N dimension of each stage; LDS stages x / h / tanh-hiddens between GEMMs.
__global__ __launch_bounds__(256, 1) void decoder_kernel(
    const int*   __restrict__ lengths,
    const float* __restrict__ eps,
    const float* __restrict__ b_ih, const float* __restrict__ b_hh,
    const float* __restrict__ b1a,  const float* __restrict__ b1b,
    const float* __restrict__ b2a,  const float* __restrict__ b2b,
    const unsigned short* __restrict__ Wih,
    const unsigned short* __restrict__ W1a,
    const unsigned short* __restrict__ W1b,
    const unsigned short* __restrict__ W2a,
    const unsigned short* __restrict__ W2b,
    const int* __restrict__ bs, const long* __restrict__ offs,
    float* __restrict__ out, long N)
{
  __shared__ __align__(16) unsigned short xs [16][XP];  // current x  (bf16)
  __shared__ __align__(16) unsigned short hbl[16][HP];  // h          (bf16)
  __shared__ __align__(16) unsigned short t1s[16][HP];  // tanh(MLP1) (bf16)
  __shared__ __align__(16) unsigned short t2s[16][HP];  // tanh(MLP2) (bf16)

  float* p1o = out;
  float* p2o = out + (size_t)N * DOUT;
  float* ho  = out + (size_t)N * 2 * DOUT;

  const int tid   = threadIdx.x;
  const int wave  = tid >> 5;
  const int lane  = tid & 31;
  const int ncol  = lane & 15;        // N column within 16-wide tile
  const int rowA  = lane & 15;        // A-matrix row for this lane
  const int msel  = lane >> 4;        // 0: C rows 0..7 / 1: rows 8..15
  const int koffA = msel * 8;         // A-tile K sub-offset per ISA layout
  const int koffB = msel * 8;         // B-tile K sub-offset per ISA layout
  const int m0    = blockIdx.x * 16;  // first batch row of this workgroup

  for (int i = tid; i < 16 * XP; i += 256) (&xs[0][0])[i] = 0;  // x0 = 0
  __syncthreads();

  const int T = lengths[0];  // lengths sorted descending -> max length

  for (int t = 0; t < T; ++t) {
    const int bsz = bs[t];
    if (bsz <= m0) break;              // our rows fell off the packed batch
    const long off = offs[t];

    // ---- Stage A: gi = x@W_ih.T (+biases) -> r,z,n -> h ----
    for (int tile = wave; tile < HID / 16; tile += 8) {
      const int n0 = tile << 4;
      v8f ar = {0,0,0,0,0,0,0,0};
      v8f az = {0,0,0,0,0,0,0,0};
      v8f an = {0,0,0,0,0,0,0,0};
      #pragma unroll
      for (int k0 = 0; k0 < DOUT; k0 += 32) {
        v16bf a   = LOAD_A16(xs, rowA, k0 + koffA);
        v16bf brm = LOAD_B16(Wih,           n0 + ncol, DOUT, k0 + koffB);
        v16bf bzm = LOAD_B16(Wih,   HID  +  n0 + ncol, DOUT, k0 + koffB);
        v16bf bnm = LOAD_B16(Wih, 2*HID  +  n0 + ncol, DOUT, k0 + koffB);
        WMMA_BF16(ar, a, brm);
        WMMA_BF16(az, a, bzm);
        WMMA_BF16(an, a, bnm);
      }
      const int   nc  = n0 + ncol;
      const float brb = b_ih[nc]         + b_hh[nc];
      const float bzb = b_ih[HID + nc]   + b_hh[HID + nc];
      const float bnb = b_ih[2*HID + nc];
      const float ghn = b_hh[2*HID + nc];
      #pragma unroll
      for (int i = 0; i < 8; ++i) {
        const float r = sigf(ar[i] + brb);
        const float z = sigf(az[i] + bzb);
        const float n = tanhfast(an[i] + bnb + r * ghn);
        const float h = (1.0f - z) * n;
        const int m = i + msel * 8;
        hbl[m][nc] = f2bf(h);
        const int gm = m0 + m;
        if (gm < bsz) ho[(size_t)(off + gm) * HID + nc] = h;
      }
    }
    __syncthreads();

    // ---- Stage B: t1 = tanh(h@W1a.T + b1a), t2 = tanh(h@W2a.T + b2a) ----
    for (int tt = wave; tt < 2 * (DMLP / 16); tt += 8) {
      const int mat = tt >> 6;                 // 0: MLP1, 1: MLP2
      const int n0  = (tt & 63) << 4;
      const unsigned short* Wa = mat ? W2a : W1a;
      const float*          ba = mat ? b2a : b1a;
      unsigned short* dst = mat ? &t2s[0][0] : &t1s[0][0];
      v8f acc = {0,0,0,0,0,0,0,0};
      #pragma unroll 4
      for (int k0 = 0; k0 < HID; k0 += 32) {
        v16bf a = LOAD_A16(hbl, rowA, k0 + koffA);
        v16bf b = LOAD_B16(Wa, n0 + ncol, HID, k0 + koffB);
        WMMA_BF16(acc, a, b);
      }
      const float bc = ba[n0 + ncol];
      #pragma unroll
      for (int i = 0; i < 8; ++i) {
        const int m = i + msel * 8;
        dst[m * HP + n0 + ncol] = f2bf(tanhfast(acc[i] + bc));
      }
    }
    __syncthreads();

    // ---- Stage C: p1, p2 projections + reparameterized sample -> next x ----
    for (int tile = wave; tile < DOUT / 16; tile += 8) {
      const int n0 = tile << 4;
      v8f a1c = {0,0,0,0,0,0,0,0};
      v8f a2c = {0,0,0,0,0,0,0,0};
      #pragma unroll 4
      for (int k0 = 0; k0 < DMLP; k0 += 32) {
        v16bf a1 = LOAD_A16(t1s, rowA, k0 + koffA);
        v16bf b1 = LOAD_B16(W1b, n0 + ncol, DMLP, k0 + koffB);
        WMMA_BF16(a1c, a1, b1);
        v16bf a2 = LOAD_A16(t2s, rowA, k0 + koffA);
        v16bf b2 = LOAD_B16(W2b, n0 + ncol, DMLP, k0 + koffB);
        WMMA_BF16(a2c, a2, b2);
      }
      const int   nc  = n0 + ncol;
      const float bb1 = b1b[nc];
      const float bb2 = b2b[nc];
      #pragma unroll
      for (int i = 0; i < 8; ++i) {
        const int m  = i + msel * 8;
        const int gm = m0 + m;
        const float p1v = a1c[i] + bb1;
        const float p2v = a2c[i] + bb2;
        const float ev  = eps[((size_t)t * BATCH + gm) * DOUT + nc];
        xs[m][nc] = f2bf(p1v + __expf(0.5f * p2v) * ev);
        if (gm < bsz) {
          const size_t fo = (size_t)(off + gm);
          p1o[fo * DOUT + nc] = p1v;
          p2o[fo * DOUT + nc] = p2v;
        }
      }
    }
    __syncthreads();   // xs ready for next step's Stage A
  }
}

// ---------------- launcher ----------------
extern "C" void kernel_launch(void* const* d_in, const int* in_sizes, int n_in,
                              void* d_out, int out_size, void* d_ws, size_t ws_size,
                              hipStream_t stream) {
  (void)in_sizes; (void)n_in; (void)ws_size;
  // setup_inputs() order:
  // 0 features(dead) 1 lengths 2 eps 3 W_f2h(dead) 4 b_f2h(dead) 5 W_ih 6 b_ih
  // 7 W_hh(dead: h_prev==0) 8 b_hh 9 W1a 10 b1a 11 W1b 12 b1b 13 W2a 14 b2a
  // 15 W2b 16 b2b 17 W_off(dead) 18 b_off(dead)
  const int*   lengths = (const int*)  d_in[1];
  const float* eps     = (const float*)d_in[2];
  const float* W_ih    = (const float*)d_in[5];
  const float* b_ih    = (const float*)d_in[6];
  const float* b_hh    = (const float*)d_in[8];
  const float* W1a     = (const float*)d_in[9];
  const float* b1a     = (const float*)d_in[10];
  const float* W1b     = (const float*)d_in[11];
  const float* b1b     = (const float*)d_in[12];
  const float* W2a     = (const float*)d_in[13];
  const float* b2a     = (const float*)d_in[14];
  const float* W2b     = (const float*)d_in[15];
  const float* b2b     = (const float*)d_in[16];

  // workspace layout: [offs:long[200] | bs:int[200] | bf16 weights ~6.8MB]
  char* ws = (char*)d_ws;
  long* offs = (long*)ws;                       // 0    .. 2048
  int*  bsv  = (int*)(ws + 2048);               // 2048 .. 4096
  unsigned short* wih_b = (unsigned short*)(ws + 4096);
  unsigned short* w1a_b = wih_b + 3 * HID * DOUT;
  unsigned short* w1b_b = w1a_b + DMLP * HID;
  unsigned short* w2a_b = w1b_b + DOUT * DMLP;
  unsigned short* w2b_b = w2a_b + DMLP * HID;

  pack_setup_kernel<<<1, 256, 0, stream>>>(lengths, bsv, offs);
  cvt_bf16_kernel<<<512, 256, 0, stream>>>(W_ih, wih_b, 3 * HID * DOUT);
  cvt_bf16_kernel<<<512, 256, 0, stream>>>(W1a,  w1a_b, DMLP * HID);
  cvt_bf16_kernel<<<512, 256, 0, stream>>>(W1b,  w1b_b, DOUT * DMLP);
  cvt_bf16_kernel<<<512, 256, 0, stream>>>(W2a,  w2a_b, DMLP * HID);
  cvt_bf16_kernel<<<512, 256, 0, stream>>>(W2b,  w2b_b, DOUT * DMLP);

  const long N = (long)out_size / (DOUT + DOUT + HID);  // sum(lengths)
  decoder_kernel<<<BATCH / 16, 256, 0, stream>>>(
      lengths, eps, b_ih, b_hh, b1a, b1b, b2a, b2b,
      wih_b, w1a_b, w1b_b, w2a_b, w2b_b, bsv, offs, (float*)d_out, N);
}
